// GRAM_27333171872172
// MI455X (gfx1250) — matrix operand order repaired and checked by reading, not compiled, not run on hardware
//
#include <hip/hip_runtime.h>
#include <hip/hip_bf16.h>

#define N_CODES 10000
#define M_ANC   32
#define DIM     256
#define ADIM    128
#define K2      (2 * DIM)   // 512: [self ; ancestor] concatenated K

typedef __attribute__((ext_vector_type(16))) __bf16 v16bf;
typedef __attribute__((ext_vector_type(8)))  float  v8f;
typedef __attribute__((ext_vector_type(4)))  float  f4;

union Frag16 { f4 f[2]; v16bf v; };

__device__ __forceinline__ int lane_id() { return (int)(threadIdx.x & 31); }

__device__ __forceinline__ int shfl_i(int v, int src) {
    return __builtin_amdgcn_ds_bpermute(src << 2, v);
}
__device__ __forceinline__ float shfl_f(float v, int src) {
    return __int_as_float(__builtin_amdgcn_ds_bpermute(src << 2, __float_as_int(v)));
}
__device__ __forceinline__ float shfl_xor_f(float v, int m) {
    return shfl_f(v, lane_id() ^ m);
}

// ---------------------------------------------------------------- prep: fp32 -> bf16
__global__ void GRAM_cvt_bf16(const float* __restrict__ src,
                              __bf16* __restrict__ dst, int n) {
    int i = blockIdx.x * blockDim.x + threadIdx.x;
    int stride = gridDim.x * blockDim.x;
    for (; i < n; i += stride) dst[i] = (__bf16)src[i];
}

// ---------------------------------------------------------------- main kernel
__global__ __launch_bounds__(256)
void GRAM_attn_kernel(const float*  __restrict__ basic,   // N x D  fp32
                      const int*    __restrict__ anc,     // N x M
                      const float*  __restrict__ maskp,   // N x M
                      const float*  __restrict__ bp,      // A
                      const float*  __restrict__ ww,      // A  (Ww[0])
                      const __bf16* __restrict__ embB,    // N x D  bf16
                      const __bf16* __restrict__ wpB,     // A x 512 bf16 (row-major Wp)
                      float*        __restrict__ out)     // N x D  fp32
{
    extern __shared__ __bf16 lws[];   // 128 KB: Wp bf16, row-major A x 512

    // Stage weights to LDS (vmem -> ds), once per block.
    {
        const f4* g = (const f4*)wpB;
        f4*       s = (f4*)lws;
        const int tot = (ADIM * K2 * 2) / 16;   // 8192 16B chunks
        for (int i = threadIdx.x; i < tot; i += blockDim.x) s[i] = g[i];
    }
    __syncthreads();

    const int lane = lane_id();
    const int h    = lane >> 4;      // half-wave: selects K sub-range in A/B layouts
    const int l16  = lane & 15;

    // Per-lane bias / attention vector entries for each N-tile: a = nt*16 + l16.
    float bpv[8], wwv[8];
#pragma unroll
    for (int nt = 0; nt < 8; ++nt) {
        bpv[nt] = bp[nt * 16 + l16];
        wwv[nt] = ww[nt * 16 + l16];
    }

    const int wavesPerBlk = blockDim.x >> 5;
    const int wave  = blockIdx.x * wavesPerBlk + (threadIdx.x >> 5);
    const int nWave = gridDim.x * wavesPerBlk;

    for (int n = wave; n < N_CODES; n += nWave) {
        // Opaque zero, re-defined every iteration by a volatile asm: every
        // B-fragment LDS read below data-depends on it, so LICM cannot hoist
        // the (loop-invariant) LDS reads out of the code loop and spill them
        // to scratch. Only the integer offset is laundered, so the compiler
        // keeps the LDS address space and still emits ds_load_b128.
        int opq = 0;
        asm volatile("" : "+v"(opq));
        const __bf16* lb = lws + opq + (size_t)l16 * K2;   // per-lane B base col

        // lane m holds ancestry entry m of this code
        int   myIdx  = anc[n * M_ANC + lane];
        float myMask = maskp[n * M_ANC + lane];

        // Row ids for the A-fragment rows of the two M-tiles (row = l16 within tile).
        int i0 = shfl_i(myIdx, l16);
        int i1 = shfl_i(myIdx, l16 + 16);
        const __bf16* r0 = embB + (size_t)i0 * DIM;
        const __bf16* r1 = embB + (size_t)i1 * DIM;
        const __bf16* sp = embB + (size_t)n  * DIM;

        v8f C[2][8];
        {
            v8f z = {0.f, 0.f, 0.f, 0.f, 0.f, 0.f, 0.f, 0.f};
#pragma unroll
            for (int mt = 0; mt < 2; ++mt)
#pragma unroll
                for (int nt = 0; nt < 8; ++nt) C[mt][nt] = z;
        }

        // ---- self half: K = 0..255. All 32 rows use the code's own embedding,
        //      so one A fragment serves both M-tiles (this realizes self_proj).
        for (int kt = 0; kt < 8; ++kt) {
            Frag16 a;
            // ISA 16-bit A layout: lane holds k = 8h+0..7 (v0-3) and 16+8h..23+8h (v4-7)
            a.f[0] = *(const f4*)(sp + kt * 32 + 8 * h);
            a.f[1] = *(const f4*)(sp + kt * 32 + 16 + 8 * h);
#pragma unroll
            for (int nt = 0; nt < 8; ++nt) {
                Frag16 b;
                // ISA 16-bit B layout: lane holds col a = nt*16+l16, k = kbase+8h..+7 and kbase+16+8h..+7
                const __bf16* bs = lb + (size_t)nt * 16 * K2 + kt * 32 + h * 8;
                b.f[0] = *(const f4*)bs;
                b.f[1] = *(const f4*)(bs + 16);
                C[0][nt] = __builtin_amdgcn_wmma_f32_16x16x32_bf16(
                    false, a.v, false, b.v, (short)0, C[0][nt], false, false);
                C[1][nt] = __builtin_amdgcn_wmma_f32_16x16x32_bf16(
                    false, a.v, false, b.v, (short)0, C[1][nt], false, false);
            }
        }

        // ---- ancestor half: K = 256..511, rows are gathered ancestor embeddings.
        for (int kt = 0; kt < 8; ++kt) {
            Frag16 a0, a1;
            a0.f[0] = *(const f4*)(r0 + kt * 32 + 8 * h);
            a0.f[1] = *(const f4*)(r0 + kt * 32 + 16 + 8 * h);
            a1.f[0] = *(const f4*)(r1 + kt * 32 + 8 * h);
            a1.f[1] = *(const f4*)(r1 + kt * 32 + 16 + 8 * h);
#pragma unroll
            for (int nt = 0; nt < 8; ++nt) {
                Frag16 b;
                const __bf16* bs = lb + (size_t)nt * 16 * K2 + DIM + kt * 32 + h * 8;
                b.f[0] = *(const f4*)bs;
                b.f[1] = *(const f4*)(bs + 16);
                C[0][nt] = __builtin_amdgcn_wmma_f32_16x16x32_bf16(
                    false, a0.v, false, b.v, (short)0, C[0][nt], false, false);
                C[1][nt] = __builtin_amdgcn_wmma_f32_16x16x32_bf16(
                    false, a1.v, false, b.v, (short)0, C[1][nt], false, false);
            }
        }

        // ---- scores[m] = sum_a tanh(C + bp) * Ww.
        // C/D layout: tile (mt,nt), reg r, lane: m = mt*16 + 8h + r, a = nt*16 + l16.
        float S[2][8];
#pragma unroll
        for (int mt = 0; mt < 2; ++mt) {
#pragma unroll
            for (int r = 0; r < 8; ++r) {
                float s = 0.f;
#pragma unroll
                for (int nt = 0; nt < 8; ++nt)
                    s += tanhf(C[mt][nt][r] + bpv[nt]) * wwv[nt];
                // reduce over the 16 lanes of this half (butterfly keeps halves apart)
                s += shfl_xor_f(s, 1);
                s += shfl_xor_f(s, 2);
                s += shfl_xor_f(s, 4);
                s += shfl_xor_f(s, 8);
                // A_att = score * mask for m = mt*16 + 8h + r
                float mk = shfl_f(myMask, mt * 16 + 8 * h + r);
                S[mt][r] = s * mk;
            }
        }

        // ---- softmax-style weights. NOTE: reference does NOT mask the weighted
        // sum; masked positions keep weight exp(0 - max). Reproduced faithfully.
        float gmax = -3.402823466e38f;
#pragma unroll
        for (int mt = 0; mt < 2; ++mt)
#pragma unroll
            for (int r = 0; r < 8; ++r) gmax = fmaxf(gmax, S[mt][r]);
        gmax = fmaxf(gmax, shfl_xor_f(gmax, 1));
        gmax = fmaxf(gmax, shfl_xor_f(gmax, 2));
        gmax = fmaxf(gmax, shfl_xor_f(gmax, 4));
        gmax = fmaxf(gmax, shfl_xor_f(gmax, 8));
        gmax = fmaxf(gmax, shfl_xor_f(gmax, 16));

        float w[2][8];
        float lsum = 0.f;
#pragma unroll
        for (int mt = 0; mt < 2; ++mt)
#pragma unroll
            for (int r = 0; r < 8; ++r) {
                w[mt][r] = __expf(S[mt][r] - gmax);
                lsum += w[mt][r];     // covers the 16 m's of this half exactly
            }
        float sumw = lsum + shfl_xor_f(lsum, 16);
        float inv  = 1.0f / sumw;

        // ---- G[n] = sum_m w_m * E_m / sumw, fp32. Lane owns d = lane*8 .. +7.
        f4 acc0 = {0.f, 0.f, 0.f, 0.f};
        f4 acc1 = {0.f, 0.f, 0.f, 0.f};
#pragma unroll
        for (int m = 0; m < 32; ++m) {
            const int mt = m >> 4, rem = m & 15, hh = rem >> 3, r = rem & 7;
            float wm = shfl_f(w[mt][r], hh * 16);   // held by lanes of half hh
            int   im = shfl_i(myIdx, m);
            const float* rp = basic + (size_t)im * DIM + lane * 8;
            f4 e0 = *(const f4*)rp;
            f4 e1 = *(const f4*)(rp + 4);
            acc0 += wm * e0;
            acc1 += wm * e1;
        }
        f4* op = (f4*)(out + (size_t)n * DIM + lane * 8);
        op[0] = acc0 * inv;
        op[1] = acc1 * inv;
    }
}

// ---------------------------------------------------------------- launcher
extern "C" void kernel_launch(void* const* d_in, const int* in_sizes, int n_in,
                              void* d_out, int out_size, void* d_ws, size_t ws_size,
                              hipStream_t stream) {
    const float* basic = (const float*)d_in[0];   // (N, D) fp32
    const int*   anc   = (const int*)  d_in[1];   // (N, M) int32
    const float* mask  = (const float*)d_in[2];   // (N, M) fp32
    const float* Wp    = (const float*)d_in[3];   // (A, 2D) fp32
    const float* bp    = (const float*)d_in[4];   // (A,)   fp32
    const float* Ww    = (const float*)d_in[5];   // (1, A) fp32
    float*       out   = (float*)d_out;           // (N, D) fp32

    // Workspace: bf16 embedding table (5,120,000 B) + bf16 Wp (131,072 B).
    __bf16* embB = (__bf16*)d_ws;
    __bf16* wpB  = (__bf16*)((char*)d_ws + (size_t)N_CODES * DIM * sizeof(__bf16));

    GRAM_cvt_bf16<<<1024, 256, 0, stream>>>(basic, embB, N_CODES * DIM);
    GRAM_cvt_bf16<<<64,   256, 0, stream>>>(Wp,    wpB,  ADIM * K2);

    // 256 blocks x 8 waves = 2048 waves; ~5 codes per wave. 128 KB dynamic LDS
    // for the Wp tile -> 2 blocks per WGP (320 KB LDS).
    GRAM_attn_kernel<<<256, 256, 131072, stream>>>(basic, anc, mask, bp, Ww,
                                                   embB, wpB, out);
}
